// DynamicLowRankAttention_41729902248007
// MI455X (gfx1250) — matrix operand order repaired and checked by reading, not compiled, not run on hardware
//
#include <hip/hip_runtime.h>
#include <hip/hip_bf16.h>
#include <math.h>

typedef __attribute__((ext_vector_type(16))) __bf16 v16bf;
typedef __attribute__((ext_vector_type(8)))  __bf16 v8bf;
typedef __attribute__((ext_vector_type(8)))  float  v8f;

#define BB   2
#define SS   2048
#define DD   1024
#define HH   16
#define HDIM 64
#define BS   4096   /* B*S  */
#define NMAT 32     /* B*H  */

static __device__ inline __bf16 f2bf(float f) {
  union { float f; unsigned u; } x; x.f = f;
  unsigned r = x.u + 0x7FFFu + ((x.u >> 16) & 1u);
  unsigned short h = (unsigned short)(r >> 16);
  __bf16 o; __builtin_memcpy(&o, &h, 2); return o;
}
static __device__ inline float bf2f(__bf16 b) {
  unsigned short h; __builtin_memcpy(&h, &b, 2);
  union { unsigned u; float f; } x; x.u = ((unsigned)h) << 16; return x.f;
}

static __device__ inline v8f wmma_bf16(v16bf a, v16bf b, v8f c) {
  return __builtin_amdgcn_wmma_f32_16x16x32_bf16(false, a, false, b, (short)0, c,
                                                 false, false);
}

// A-operand (16x32 bf16, wave32): lane holds row (lane&15).
// elements 0..7  = K in [hf*8, hf*8+8), elements 8..15 = K in [16+hf*8, 16+hf*8+8)
static __device__ inline v16bf load_a(const __bf16* row, int hf) {
  v8bf lo = *(const v8bf*)(row + hf * 8);
  v8bf hi = *(const v8bf*)(row + 16 + hf * 8);
  return __builtin_shufflevector(lo, hi, 0,1,2,3,4,5,6,7,8,9,10,11,12,13,14,15);
}
// B-operand (32x16 bf16): lane holds column (lane&15); elements 0..7 = K in [hf*8, hf*8+8), 8..15 = K in [16+hf*8, 16+hf*8+8).
// colrow points at K=0 of this lane's column (a row of the K-major/transposed storage).
static __device__ inline v16bf load_b(const __bf16* colrow, int hf) {
  v8bf lo = *(const v8bf*)(colrow + hf * 8);
  v8bf hi = *(const v8bf*)(colrow + 16 + hf * 8);
  return __builtin_shufflevector(lo, hi, 0,1,2,3,4,5,6,7,8,9,10,11,12,13,14,15);
}

/* ---------------- stage 0: precision conversion ---------------- */

__global__ void k_cvt_x(const float* __restrict__ x, __bf16* __restrict__ xb, int n) {
  int i = blockIdx.x * blockDim.x + threadIdx.x;
  if (i < n) xb[i] = f2bf(x[i]);
}

// Wt[n][k] = W[k][n]  (bf16)
__global__ void k_wt(const float* __restrict__ w, __bf16* __restrict__ wt) {
  int i = blockIdx.x * blockDim.x + threadIdx.x;   // i = n*1024 + k
  int n = i >> 10, k = i & 1023;
  wt[i] = f2bf(w[k * DD + n]);
}

/* ---------------- stage 1: fused QKV projection (WMMA, ping-pong) ---------- */

__global__ __launch_bounds__(256) void k_qkv(
    const __bf16* __restrict__ xb,
    const __bf16* __restrict__ wqt, const __bf16* __restrict__ wkt,
    const __bf16* __restrict__ wvt,
    const float* __restrict__ bq, const float* __restrict__ bk,
    const float* __restrict__ bv,
    __bf16* __restrict__ Qb, __bf16* __restrict__ Kb, __bf16* __restrict__ Vtb) {
  const int z = blockIdx.z;
  const __bf16* wt  = (z == 0) ? wqt : (z == 1) ? wkt : wvt;
  const float*  bia = (z == 0) ? bq  : (z == 1) ? bk  : bv;
  int w    = (blockIdx.x * blockDim.x + threadIdx.x) >> 5;
  int lane = threadIdx.x & 31;
  int col  = lane & 15, hf = lane >> 4;
  int mtile = w >> 4, ntile = w & 15;

  const __bf16* arow = xb + (size_t)(mtile * 16 + col) * DD;
  const __bf16* brow[4];
#pragma unroll
  for (int t = 0; t < 4; t++) brow[t] = wt + (size_t)(ntile * 64 + t * 16 + col) * DD;

  v8f acc[4] = {v8f{}, v8f{}, v8f{}, v8f{}};
  // ping-pong buffers: no register rotation, loads land in-place.
  v16bf aP = load_a(arow, hf);
  v16bf bP[4];
#pragma unroll
  for (int t = 0; t < 4; t++) bP[t] = load_b(brow[t], hf);
  v16bf aQ;
  v16bf bQ[4];
  for (int k0 = 0; k0 < DD; k0 += 64) {
    aQ = load_a(arow + k0 + 32, hf);
#pragma unroll
    for (int t = 0; t < 4; t++) bQ[t] = load_b(brow[t] + k0 + 32, hf);
#pragma unroll
    for (int t = 0; t < 4; t++) acc[t] = wmma_bf16(aP, bP[t], acc[t]);
    if (k0 + 64 < DD) {
      aP = load_a(arow + k0 + 64, hf);
#pragma unroll
      for (int t = 0; t < 4; t++) bP[t] = load_b(brow[t] + k0 + 64, hf);
    }
#pragma unroll
    for (int t = 0; t < 4; t++) acc[t] = wmma_bf16(aQ, bQ[t], acc[t]);
  }

  float bv4[4];
#pragma unroll
  for (int t = 0; t < 4; t++) bv4[t] = bia[ntile * 64 + t * 16 + col];

  int b = (mtile * 16) >> 11;          // batch index
  int sbase = (mtile * 16) & 2047;     // sequence base of this tile
  if (z < 2) {
    __bf16* mbase = ((z == 0) ? Qb : Kb) + (size_t)(b * HH + ntile) * SS * HDIM;
#pragma unroll
    for (int t = 0; t < 4; t++)
#pragma unroll
      for (int r = 0; r < 8; r++) {
        int s = sbase + r + 8 * hf;
        mbase[(size_t)s * HDIM + t * 16 + col] = f2bf(acc[t][r] + bv4[t]);
      }
  } else {  // V stored transposed: [b,h,d,s]
    __bf16* mbase = Vtb + (size_t)(b * HH + ntile) * HDIM * SS;
#pragma unroll
    for (int t = 0; t < 4; t++) {
      int d = t * 16 + col;
      v8bf vv;
#pragma unroll
      for (int r = 0; r < 8; r++) vv[r] = f2bf(acc[t][r] + bv4[t]);
      *(v8bf*)(mbase + (size_t)d * SS + sbase + 8 * hf) = vv;
    }
  }
}

/* ---------------- stage 2: Gram matrices G = A^T A (64x64 per head) ------- */

__global__ __launch_bounds__(256) void k_gram(const __bf16* __restrict__ Qb,
                                              const __bf16* __restrict__ Kb,
                                              float* __restrict__ G) {
  __shared__ __bf16 tile[64 * 64];
  int blk = blockIdx.x;  // 0..63 : [0,32)=Q heads, [32,64)=K heads
  const __bf16* A = ((blk >> 5) == 0 ? Qb : Kb) + (size_t)(blk & 31) * SS * HDIM;
  int t = threadIdx.x;
  int i = t >> 2, j0 = (t & 3) * 16;
  float acc[16];
#pragma unroll
  for (int jj = 0; jj < 16; jj++) acc[jj] = 0.f;

  for (int s0 = 0; s0 < SS; s0 += 64) {
    *(v16bf*)(tile + (t >> 2) * 64 + (t & 3) * 16) =
        *(const v16bf*)(A + (size_t)(s0 + (t >> 2)) * 64 + (t & 3) * 16);
    __syncthreads();
    for (int s = 0; s < 64; s++) {
      float ai = bf2f(tile[s * 64 + i]);
#pragma unroll
      for (int jj = 0; jj < 16; jj++) acc[jj] += ai * bf2f(tile[s * 64 + j0 + jj]);
    }
    __syncthreads();
  }
  float* g = G + (size_t)blk * 4096 + i * 64 + j0;
#pragma unroll
  for (int jj = 0; jj < 16; jj++) g[jj] = acc[jj];
}

/* ---------------- stage 3: rank-16 projector P = Y Y^T via subspace iter --- */

__global__ __launch_bounds__(64) void k_proj(const float* __restrict__ G,
                                             __bf16* __restrict__ Pb) {
  __shared__ float Gs[64 * 64];
  __shared__ float Ys[64 * 16];
  __shared__ float Zs[64 * 16];
  __shared__ float Cs[16 * 16];
  int i = threadIdx.x;
  const float* g = G + (size_t)blockIdx.x * 4096;
  for (int idx = i; idx < 4096; idx += 64) Gs[idx] = g[idx];
  __syncthreads();
  for (int j = 0; j < 16; j++)
    Ys[i * 16 + j] = Gs[i * 64 + j] + ((i == j) ? 1.0f : 0.0f);
  __syncthreads();

  for (int it = 0; it < 8; ++it) {
    // Z = G * Y
    float z[16];
#pragma unroll
    for (int j = 0; j < 16; j++) z[j] = 0.f;
    for (int k = 0; k < 64; k++) {
      float gg = Gs[i * 64 + k];
#pragma unroll
      for (int j = 0; j < 16; j++) z[j] += gg * Ys[k * 16 + j];
    }
#pragma unroll
    for (int j = 0; j < 16; j++) Zs[i * 16 + j] = z[j];
    __syncthreads();
    // C = Z^T Z (4 entries / thread)
#pragma unroll
    for (int e = 0; e < 4; e++) {
      int idx = i * 4 + e, p = idx >> 4, q = idx & 15;
      float c = 0.f;
      for (int k = 0; k < 64; k++) c += Zs[k * 16 + p] * Zs[k * 16 + q];
      Cs[idx] = c;
    }
    __syncthreads();
    float tr = 0.f;
    for (int p = 0; p < 16; p++) tr += Cs[p * 17];
    float sc = 16.0f / (tr + 1e-30f), sq = sqrtf(sc);
    __syncthreads();
    for (int j = 0; j < 16; j++) Zs[i * 16 + j] *= sq;
#pragma unroll
    for (int e = 0; e < 4; e++) Cs[i * 4 + e] *= sc;
    __syncthreads();
    // Newton-Schulz 1: Y = Z (1.5 I - 0.5 C)
    for (int p = 0; p < 16; p++) {
      float a = 1.5f * Zs[i * 16 + p];
      for (int q = 0; q < 16; q++) a -= 0.5f * Zs[i * 16 + q] * Cs[q * 16 + p];
      Ys[i * 16 + p] = a;
    }
    __syncthreads();
    // C = Y^T Y
#pragma unroll
    for (int e = 0; e < 4; e++) {
      int idx = i * 4 + e, p = idx >> 4, q = idx & 15;
      float c = 0.f;
      for (int k = 0; k < 64; k++) c += Ys[k * 16 + p] * Ys[k * 16 + q];
      Cs[idx] = c;
    }
    __syncthreads();
    // Newton-Schulz 2 (row-private in-place)
    float y2[16];
    for (int p = 0; p < 16; p++) {
      float a = 1.5f * Ys[i * 16 + p];
      for (int q = 0; q < 16; q++) a -= 0.5f * Ys[i * 16 + q] * Cs[q * 16 + p];
      y2[p] = a;
    }
#pragma unroll
    for (int p = 0; p < 16; p++) Ys[i * 16 + p] = y2[p];
    __syncthreads();
  }
  // P row i = Y[i,:] * Y^T  (P symmetric)
  for (int j = 0; j < 64; j++) {
    float a = 0.f;
#pragma unroll
    for (int q = 0; q < 16; q++) a += Ys[i * 16 + q] * Ys[j * 16 + q];
    Pb[(size_t)blockIdx.x * 4096 + i * 64 + j] = f2bf(a);
  }
}

/* ---------------- stage 4: apply projector, Q <- Q P, K <- K P (WMMA) ------ */

__global__ __launch_bounds__(256) void k_apply(__bf16* __restrict__ Qb,
                                               __bf16* __restrict__ Kb,
                                               const __bf16* __restrict__ Pb) {
  int w = (blockIdx.x * blockDim.x + threadIdx.x) >> 5;
  int lane = threadIdx.x & 31, col = lane & 15, hf = lane >> 4;
  int matsel = w >> 12;
  int rem = w & 4095;
  int mat = rem >> 7, mtile = rem & 127;
  __bf16* base = (matsel == 0 ? Qb : Kb) + (size_t)mat * SS * HDIM;
  const __bf16* P = Pb + (size_t)(matsel * NMAT + mat) * 4096;
  const __bf16* arow = base + (size_t)(mtile * 16 + col) * HDIM;
  const __bf16* brow[4];
#pragma unroll
  for (int t = 0; t < 4; t++) brow[t] = P + (size_t)(t * 16 + col) * 64;

  v8f acc[4] = {v8f{}, v8f{}, v8f{}, v8f{}};
#pragma unroll
  for (int k0 = 0; k0 < 64; k0 += 32) {
    v16bf a = load_a(arow + k0, hf);
#pragma unroll
    for (int t = 0; t < 4; t++)
      acc[t] = wmma_bf16(a, load_b(brow[t] + k0, hf), acc[t]);
  }
#pragma unroll
  for (int t = 0; t < 4; t++)
#pragma unroll
    for (int r = 0; r < 8; r++)
      base[(size_t)(mtile * 16 + r + 8 * hf) * HDIM + t * 16 + col] = f2bf(acc[t][r]);
}

/* ---------------- stage 5: flash attention (WMMA + online softmax) --------- */

__global__ __launch_bounds__(256) void k_attn(const __bf16* __restrict__ Qb,
                                              const __bf16* __restrict__ Kb,
                                              const __bf16* __restrict__ Vtb,
                                              __bf16* __restrict__ ctxb) {
  __shared__ __bf16 pt[8 * 512];  // per-wave 16x32 bf16 probability tile
  int w = (blockIdx.x * blockDim.x + threadIdx.x) >> 5;
  int lane = threadIdx.x & 31, col = lane & 15, hf = lane >> 4;
  __bf16* myp = pt + (threadIdx.x >> 5) * 512;
  int mat = w >> 7, qtile = w & 127;

  const __bf16* Qm = Qb + (size_t)mat * SS * HDIM;
  const __bf16* Km = Kb + (size_t)mat * SS * HDIM;
  const __bf16* Vm = Vtb + (size_t)mat * HDIM * SS;

  const __bf16* qrow = Qm + (size_t)(qtile * 16 + col) * HDIM;
  v16bf qa0 = load_a(qrow, hf);
  v16bf qa1 = load_a(qrow + 32, hf);
  const __bf16* vrow[4];
#pragma unroll
  for (int t = 0; t < 4; t++) vrow[t] = Vm + (size_t)(t * 16 + col) * SS;

  v8f acc[4] = {v8f{}, v8f{}, v8f{}, v8f{}};
  float M8[8], L8[8];
#pragma unroll
  for (int r = 0; r < 8; r++) { M8[r] = -1e30f; L8[r] = 0.f; }

  // prefetch K-block 0 B-operands
  const __bf16* k0row = Km + (size_t)col * HDIM;
  const __bf16* k1row = Km + (size_t)(16 + col) * HDIM;
  v16bf kb00 = load_b(k0row, hf);
  v16bf kb01 = load_b(k0row + 32, hf);
  v16bf kb10 = load_b(k1row, hf);
  v16bf kb11 = load_b(k1row + 32, hf);

  for (int kb = 0; kb < SS; kb += 32) {
    v8f s0 = {};
    s0 = wmma_bf16(qa0, kb00, s0);
    s0 = wmma_bf16(qa1, kb01, s0);
    v8f s1 = {};
    s1 = wmma_bf16(qa0, kb10, s1);
    s1 = wmma_bf16(qa1, kb11, s1);

    // issue V loads (current block) and K loads (next block) early: they
    // overlap with the softmax VALU chain and the LDS transpose below.
    v16bf vb0 = load_b(vrow[0] + kb, hf);
    v16bf vb1 = load_b(vrow[1] + kb, hf);
    v16bf vb2 = load_b(vrow[2] + kb, hf);
    v16bf vb3 = load_b(vrow[3] + kb, hf);
    if (kb + 32 < SS) {
      const __bf16* n0 = Km + (size_t)(kb + 32 + col) * HDIM;
      const __bf16* n1 = Km + (size_t)(kb + 48 + col) * HDIM;
      kb00 = load_b(n0, hf);
      kb01 = load_b(n0 + 32, hf);
      kb10 = load_b(n1, hf);
      kb11 = load_b(n1 + 32, hf);
    }

#pragma unroll
    for (int r = 0; r < 8; r++) {
      float a0 = s0[r] * 0.125f, a1 = s1[r] * 0.125f;  // 1/sqrt(64)
      float mx = fmaxf(a0, a1);
#pragma unroll
      for (int m = 1; m < 16; m <<= 1) mx = fmaxf(mx, __shfl_xor(mx, m, 32));
      float nm = fmaxf(M8[r], mx);
      float al = __expf(M8[r] - nm);
      M8[r] = nm;
      float p0 = __expf(a0 - nm), p1 = __expf(a1 - nm);
      float rs = p0 + p1;
#pragma unroll
      for (int m = 1; m < 16; m <<= 1) rs += __shfl_xor(rs, m, 32);
      L8[r] = L8[r] * al + rs;
      acc[0][r] *= al; acc[1][r] *= al; acc[2][r] *= al; acc[3][r] *= al;
      int mrow = r + 8 * hf;
      myp[mrow * 32 + col]      = f2bf(p0);
      myp[mrow * 32 + 16 + col] = f2bf(p1);
    }
    asm volatile("s_wait_dscnt 0" ::: "memory");  // intra-wave LDS RAW
    v16bf pa = load_a(myp + col * 32, hf);
    acc[0] = wmma_bf16(pa, vb0, acc[0]);
    acc[1] = wmma_bf16(pa, vb1, acc[1]);
    acc[2] = wmma_bf16(pa, vb2, acc[2]);
    acc[3] = wmma_bf16(pa, vb3, acc[3]);
  }

  int b = mat >> 4, h = mat & 15;
#pragma unroll
  for (int t = 0; t < 4; t++)
#pragma unroll
    for (int r = 0; r < 8; r++) {
      int s = qtile * 16 + r + 8 * hf;
      ctxb[(size_t)(b * SS + s) * DD + h * 64 + t * 16 + col] =
          f2bf(acc[t][r] / L8[r]);
    }
}

/* ---------------- stage 6: output projection (WMMA, fp32 out, ping-pong) --- */

__global__ __launch_bounds__(256) void k_out(const __bf16* __restrict__ ctxb,
                                             const __bf16* __restrict__ wot,
                                             const float* __restrict__ bo,
                                             float* __restrict__ out) {
  int w = (blockIdx.x * blockDim.x + threadIdx.x) >> 5;
  int lane = threadIdx.x & 31, col = lane & 15, hf = lane >> 4;
  int mtile = w >> 4, ntile = w & 15;

  const __bf16* arow = ctxb + (size_t)(mtile * 16 + col) * DD;
  const __bf16* brow[4];
#pragma unroll
  for (int t = 0; t < 4; t++) brow[t] = wot + (size_t)(ntile * 64 + t * 16 + col) * DD;

  v8f acc[4] = {v8f{}, v8f{}, v8f{}, v8f{}};
  v16bf aP = load_a(arow, hf);
  v16bf bP[4];
#pragma unroll
  for (int t = 0; t < 4; t++) bP[t] = load_b(brow[t], hf);
  v16bf aQ;
  v16bf bQ[4];
  for (int k0 = 0; k0 < DD; k0 += 64) {
    aQ = load_a(arow + k0 + 32, hf);
#pragma unroll
    for (int t = 0; t < 4; t++) bQ[t] = load_b(brow[t] + k0 + 32, hf);
#pragma unroll
    for (int t = 0; t < 4; t++) acc[t] = wmma_bf16(aP, bP[t], acc[t]);
    if (k0 + 64 < DD) {
      aP = load_a(arow + k0 + 64, hf);
#pragma unroll
      for (int t = 0; t < 4; t++) bP[t] = load_b(brow[t] + k0 + 64, hf);
    }
#pragma unroll
    for (int t = 0; t < 4; t++) acc[t] = wmma_bf16(aQ, bQ[t], acc[t]);
  }

#pragma unroll
  for (int t = 0; t < 4; t++) {
    float bias = bo[ntile * 64 + t * 16 + col];
#pragma unroll
    for (int r = 0; r < 8; r++)
      out[(size_t)(mtile * 16 + r + 8 * hf) * DD + ntile * 64 + t * 16 + col] =
          acc[t][r] + bias;
  }
}

/* ---------------- host launch ---------------- */

extern "C" void kernel_launch(void* const* d_in, const int* in_sizes, int n_in,
                              void* d_out, int out_size, void* d_ws, size_t ws_size,
                              hipStream_t stream) {
  const float* x  = (const float*)d_in[0];
  const float* Wq = (const float*)d_in[1];
  const float* bq = (const float*)d_in[2];
  const float* Wk = (const float*)d_in[3];
  const float* bk = (const float*)d_in[4];
  const float* Wv = (const float*)d_in[5];
  const float* bv = (const float*)d_in[6];
  const float* Wo = (const float*)d_in[7];
  const float* bo = (const float*)d_in[8];
  float* out = (float*)d_out;

  char* ws = (char*)d_ws;
  __bf16* Xb   = (__bf16*)(ws + 0);          // 8 MB
  __bf16* Wqt  = (__bf16*)(ws + 8388608);    // 2 MB each
  __bf16* Wkt  = (__bf16*)(ws + 10485760);
  __bf16* Wvt  = (__bf16*)(ws + 12582912);
  __bf16* Wot  = (__bf16*)(ws + 14680064);
  __bf16* Qb   = (__bf16*)(ws + 16777216);   // 8 MB
  __bf16* Kb   = (__bf16*)(ws + 25165824);   // 8 MB
  __bf16* Vtb  = (__bf16*)(ws + 33554432);   // 8 MB
  __bf16* Ctxb = (__bf16*)(ws + 41943040);   // 8 MB
  float*  G    = (float* )(ws + 50331648);   // 1 MB
  __bf16* Pb   = (__bf16*)(ws + 51380224);   // 0.5 MB

  k_cvt_x<<<16384, 256, 0, stream>>>(x, Xb, BS * DD);
  k_wt<<<4096, 256, 0, stream>>>(Wq, Wqt);
  k_wt<<<4096, 256, 0, stream>>>(Wk, Wkt);
  k_wt<<<4096, 256, 0, stream>>>(Wv, Wvt);
  k_wt<<<4096, 256, 0, stream>>>(Wo, Wot);

  dim3 g1(512, 1, 3);
  k_qkv<<<g1, 256, 0, stream>>>(Xb, Wqt, Wkt, Wvt, bq, bk, bv, Qb, Kb, Vtb);

  k_gram<<<64, 256, 0, stream>>>(Qb, Kb, G);
  k_proj<<<64, 64, 0, stream>>>(G, Pb);
  k_apply<<<1024, 256, 0, stream>>>(Qb, Kb, Pb);

  k_attn<<<512, 256, 0, stream>>>(Qb, Kb, Vtb, Ctxb);
  k_out<<<512, 256, 0, stream>>>(Ctxb, Wot, bo, out);
}
